// SAGE_SuperGAT_12498354831545
// MI455X (gfx1250) — compile-verified, matrix operand order checked by reading, not run on hardware
//
#include <hip/hip_runtime.h>
#include <hip/hip_bf16.h>

// ---------------------------------------------------------------------------
// CDNA5 (gfx1250) two-branch SAGE/SuperGAT GNN.
//  * Dense layers: v_wmma_f32_16x16x32_bf16, wave32, NB (2 or 4) 16x16 output
//    tiles per wave sharing one A fragment.
//    - BOTH operands pre-packed to bf16 in exact WMMA fragment layouts:
//      A: [mtile][kb][lane*16+j]  (pack_a_kernel, handles K padding)
//      B: [kb][nb][lane*16+j]     (pack_w_kernel; NB tiles contiguous per kb)
//    - GEMM inner loop: 2 b128 A loads + 2*NB b128 B loads (immediate
//      offsets, one base pointer each) + NB chained WMMAs. No converts,
//      no bounds checks, no flat loads.
//  * Graph aggregation / attention softmax: f32 atomics + wave-shuffle
//    reductions (bandwidth bound; activation panels largely fit in 192MB L2).
// ---------------------------------------------------------------------------

typedef __attribute__((ext_vector_type(16))) __bf16 v16bf;
typedef __attribute__((ext_vector_type(8)))  float  v8f;

#define NUM_GRAPHS 512

// ---------------- helpers ----------------
__device__ __forceinline__ void atomicMaxFloat(float* addr, float val) {
  unsigned int* ua = (unsigned int*)addr;
  unsigned int old = *ua;
  while (true) {
    float f = __builtin_bit_cast(float, old);
    if (f >= val) break;
    unsigned int assumed = old;
    old = atomicCAS(ua, assumed, __builtin_bit_cast(unsigned int, val));
    if (old == assumed) break;
  }
}

// ---------------- utility kernels ----------------
__global__ void fill_kernel(float* __restrict__ p, float v, long n) {
  long i = (long)blockIdx.x * blockDim.x + threadIdx.x;
  if (i < n) p[i] = v;
}

// dst[Ks] = src[K] zero-padded (used for biases)
__global__ void pad_copy_kernel(const float* __restrict__ src, float* __restrict__ dst,
                                long M, int K, int Ks) {
  long i = (long)blockIdx.x * blockDim.x + threadIdx.x;
  if (i >= M * (long)Ks) return;
  long r = i / Ks; int c = (int)(i % Ks);
  dst[i] = (c < K) ? src[r * K + c] : 0.0f;
}

// f32 A[M x K] (row stride K) -> bf16 packed A-fragment layout:
// Ap[mt][kb][lane*16+j], m = mt*16 + (lane&15),
//                        k = kb*32 + (j>=8?16:0) + (lane>>4)*8 + (j&7).
// k >= K -> 0 (handles padding from unpadded sources).
__global__ void pack_a_kernel(const float* __restrict__ A, __bf16* __restrict__ Ap,
                              int K, int Ks, long total) {
  long idx = (long)blockIdx.x * blockDim.x + threadIdx.x;
  if (idx >= total) return;
  int j    = (int)(idx & 15);
  int lane = (int)((idx >> 4) & 31);
  long blk = idx >> 9;
  int nkb = Ks >> 5;
  long kb = blk % nkb;
  long mt = blk / nkb;
  int  k = (int)(kb << 5) + ((j >= 8) ? 16 : 0) + ((lane >> 4) << 3) + (j & 7);
  long m = (mt << 4) + (lane & 15);
  float v = (k < K) ? A[m * K + k] : 0.0f;
  Ap[idx] = (__bf16)v;
}

// f32 W[K x N] -> bf16 packed B-fragment layout (kb-major so the NB tiles of
// one k-step are contiguous):
// Wp[kb][nb][lane*16+j], k = kb*32 + (j>=8?16:0) + (lane>>4)*8 + (j&7),
//                        n = nb*16 + (lane&15).  OOB -> 0.
__global__ void pack_w_kernel(const float* __restrict__ W, __bf16* __restrict__ Wp,
                              int K, int N, int Ks, int Ns, long total) {
  long idx = (long)blockIdx.x * blockDim.x + threadIdx.x;
  if (idx >= total) return;
  int j    = (int)(idx & 15);
  int lane = (int)((idx >> 4) & 31);
  long blk = idx >> 9;
  int tilesN = Ns >> 4;
  int nb = (int)(blk % tilesN);
  int kb = (int)(blk / tilesN);
  int k = (kb << 5) + ((j >= 8) ? 16 : 0) + ((lane >> 4) << 3) + (j & 7);
  int n = (nb << 4) + (lane & 15);
  float v = (k < K && n < N) ? W[(long)k * N + n] : 0.0f;
  Wp[idx] = (__bf16)v;
}

__global__ void degree_kernel(const int* __restrict__ dst, int E, float* __restrict__ deg) {
  int e = blockIdx.x * blockDim.x + threadIdx.x;
  if (e < E) atomicAdd(&deg[dst[e]], 1.0f);
}

// block per edge, threads stride feature dim; separate src/dst strides
__global__ void edge_scatter_add_kernel(const float* __restrict__ x,
                                        const int* __restrict__ src,
                                        const int* __restrict__ dst,
                                        int E, int C, int CsSrc, int CsDst,
                                        float* __restrict__ acc) {
  int e = blockIdx.x;
  if (e >= E) return;
  long s = src[e], d = dst[e];
  for (int c = threadIdx.x; c < C; c += blockDim.x)
    atomicAdd(&acc[d * CsDst + c], x[s * CsSrc + c]);
}

__global__ void divide_by_deg_kernel(float* __restrict__ acc, const float* __restrict__ deg,
                                     long Nn, int C, int Cs) {
  long i = (long)blockIdx.x * blockDim.x + threadIdx.x;
  if (i >= Nn * (long)C) return;
  long r = i / C; int c = (int)(i % C);
  acc[r * Cs + c] = acc[r * Cs + c] / fmaxf(deg[r], 1.0f);
}

// ---------------- WMMA GEMM (all-bf16 packed operands) ----------------
template <int NB>
__device__ __forceinline__ void gemm_loop_packed(v8f* acc,
                                                 const __bf16* __restrict__ Ap,
                                                 const __bf16* __restrict__ Wp,
                                                 int mt, int nkb, int tilesN,
                                                 int nb0, int lane) {
  const __bf16* ap = Ap + ((long)mt * nkb) * 512 + lane * 16;
  const __bf16* bp = Wp + (long)nb0 * 512 + lane * 16;
  const long bstep = (long)tilesN * 512;
  for (int kb = 0; kb < nkb; ++kb) {
    v16bf a = *(const v16bf*)ap;
#pragma unroll
    for (int q = 0; q < NB; ++q) {
      v16bf b = *(const v16bf*)(bp + q * 512);   // constant byte offsets
      acc[q] = __builtin_amdgcn_wmma_f32_16x16x32_bf16(false, a, false, b,
                                                       (short)0, acc[q], false, false);
    }
    ap += 512;
    bp += bstep;
  }
}

// out[M x Ns] = act(Ap @ Wp [+ Ap2 @ Wp2] + biasp).
// M % 16 == 0, Ks/Ks2 % 32 == 0, (Ns/16) % NB == 0; biasp has Ns entries.
template <int NB>
__global__ void __launch_bounds__(256)
gemm_wmma_kernel(const __bf16* __restrict__ Ap,  const __bf16* __restrict__ Wp,
                 const __bf16* __restrict__ Ap2, const __bf16* __restrict__ Wp2,
                 const float* __restrict__ biasp, float* __restrict__ out,
                 int M, int Ns, int Ks, int Ks2, int do_relu) {
  const int lane = threadIdx.x & 31;
  const int wave = threadIdx.x >> 5;
  const int tilesN = Ns >> 4;
  const int nGroups = tilesN / NB;
  const int tile = blockIdx.x * 8 + wave;
  if (tile >= (M >> 4) * nGroups) return;       // wave-uniform
  const int mt  = tile / nGroups;
  const int nb0 = (tile % nGroups) * NB;
  const int half = lane >> 4;
  const int l15  = lane & 15;
  const int m0   = mt << 4;

  v8f acc[NB];
#pragma unroll
  for (int q = 0; q < NB; ++q) acc[q] = v8f{};

  gemm_loop_packed<NB>(acc, Ap, Wp, mt, Ks >> 5, tilesN, nb0, lane);
  if (Ap2 != nullptr)
    gemm_loop_packed<NB>(acc, Ap2, Wp2, mt, Ks2 >> 5, tilesN, nb0, lane);

#pragma unroll
  for (int q = 0; q < NB; ++q) {
    const int ncol = ((nb0 + q) << 4) + l15;
    const float bn = biasp[ncol];
#pragma unroll
    for (int r = 0; r < 8; ++r) {               // D layout: VGPR r -> M = m0+8*half+r
      float v = acc[q][r] + bn;
      if (do_relu) v = fmaxf(v, 0.0f);
      out[(long)(m0 + half * 8 + r) * Ns + ncol] = v;
    }
  }
}

// ---------------- SuperGAT attention ----------------
__global__ void node_att_kernel(const float* __restrict__ h, int Nn, int C, int Cs,
                                const float* __restrict__ attl, const float* __restrict__ attr,
                                float* __restrict__ al, float* __restrict__ ar) {
  int i = blockIdx.x * (blockDim.x >> 5) + (threadIdx.x >> 5);
  int lane = threadIdx.x & 31;
  if (i >= Nn) return;
  float pl = 0.0f, pr = 0.0f;
  for (int c = lane; c < C; c += 32) {
    float hv = h[(long)i * Cs + c];
    pl += hv * attl[c];
    pr += hv * attr[c];
  }
  for (int off = 16; off > 0; off >>= 1) {
    pl += __shfl_down(pl, off);
    pr += __shfl_down(pr, off);
  }
  if (lane == 0) { al[i] = pl; ar[i] = pr; }
}

// pass 1: alpha = leaky_relu((al[s]+ar[d]) * sigmoid(<h_d,h_s>)); segment max.
// edges E..E+Nn-1 are self-loops; one wave per edge.
__global__ void edge_alpha_kernel(const float* __restrict__ h,
                                  const int* __restrict__ src, const int* __restrict__ dst,
                                  int E, int Nn, int C, int Cs,
                                  const float* __restrict__ al, const float* __restrict__ ar,
                                  float* __restrict__ alpha, float* __restrict__ mmax) {
  int e = blockIdx.x * (blockDim.x >> 5) + (threadIdx.x >> 5);
  int lane = threadIdx.x & 31;
  int total = E + Nn;
  if (e >= total) return;
  int s = (e < E) ? src[e] : (e - E);
  int d = (e < E) ? dst[e] : (e - E);
  float part = 0.0f;
  for (int c = lane; c < C; c += 32)
    part += h[(long)d * Cs + c] * h[(long)s * Cs + c];
  for (int off = 16; off > 0; off >>= 1) part += __shfl_down(part, off);
  if (lane == 0) {
    float sg = 1.0f / (1.0f + __expf(-part));
    float a = (al[s] + ar[d]) * sg;
    a = (a > 0.0f) ? a : 0.2f * a;              // leaky_relu(0.2)
    alpha[e] = a;
    atomicMaxFloat(&mmax[d], a);
  }
}

__global__ void edge_exp_kernel(const int* __restrict__ src, const int* __restrict__ dst,
                                int E, int Nn, float* __restrict__ alpha,
                                const float* __restrict__ mmax, float* __restrict__ denom) {
  int e = blockIdx.x * blockDim.x + threadIdx.x;
  if (e >= E + Nn) return;
  int d = (e < E) ? dst[e] : (e - E);
  float ev = __expf(alpha[e] - mmax[d]);
  alpha[e] = ev;
  atomicAdd(&denom[d], ev);
}

__global__ void edge_aggregate_kernel(const float* __restrict__ h,
                                      const int* __restrict__ src, const int* __restrict__ dst,
                                      int E, int Nn, int C, int Cs,
                                      const float* __restrict__ alpha,
                                      const float* __restrict__ denom,
                                      float* __restrict__ outacc) {
  int e = blockIdx.x;
  if (e >= E + Nn) return;
  long s = (e < E) ? src[e] : (e - E);
  long d = (e < E) ? dst[e] : (e - E);
  float w = alpha[e] / denom[d];                // denom > 0 (self-loop included)
  for (int c = threadIdx.x; c < C; c += blockDim.x)
    atomicAdd(&outacc[d * Cs + c], h[s * Cs + c] * w);
}

__global__ void bias_relu_kernel(float* __restrict__ x, const float* __restrict__ b,
                                 long Nn, int C, int Cs) {
  long i = (long)blockIdx.x * blockDim.x + threadIdx.x;
  if (i >= Nn * (long)C) return;
  long r = i / C; int c = (int)(i % C);
  x[r * Cs + c] = fmaxf(x[r * Cs + c] + b[c], 0.0f);
}

// ---------------- pooling / concat / final copy ----------------
__global__ void pool_add_kernel(const float* __restrict__ x, const int* __restrict__ batch,
                                int Nn, int C, int Cs,
                                float* __restrict__ gsum, float* __restrict__ gcnt) {
  int i = blockIdx.x;
  if (i >= Nn) return;
  long g = batch[i];
  for (int c = threadIdx.x; c < C; c += blockDim.x)
    atomicAdd(&gsum[g * Cs + c], x[(long)i * Cs + c]);
  if (threadIdx.x == 0) atomicAdd(&gcnt[g], 1.0f);
}

__global__ void pool_div_kernel(float* __restrict__ gsum, const float* __restrict__ gcnt,
                                int C, int Cs) {
  long i = (long)blockIdx.x * blockDim.x + threadIdx.x;
  if (i >= (long)NUM_GRAPHS * C) return;
  long r = i / C; int c = (int)(i % C);
  gsum[r * Cs + c] = gsum[r * Cs + c] / fmaxf(gcnt[r], 1.0f);
}

__global__ void concat_kernel(const float* __restrict__ a, const float* __restrict__ b,
                              float* __restrict__ cat, int G, int F) {
  int i = blockIdx.x * blockDim.x + threadIdx.x;
  if (i >= G * F) return;
  int g = i / F, f = i % F;
  cat[(long)g * 2 * F + f]     = a[i];
  cat[(long)g * 2 * F + F + f] = b[i];
}

__global__ void copy_col0_kernel(const float* __restrict__ src, float* __restrict__ dst,
                                 int G, int Ns) {
  int i = blockIdx.x * blockDim.x + threadIdx.x;
  if (i < G) dst[i] = src[(long)i * Ns];
}

// ---------------- host-side orchestration ----------------
static inline int cdiv(long a, long b) { return (int)((a + b - 1) / b); }

static void launch_fill(float* p, float v, long n, hipStream_t s) {
  fill_kernel<<<cdiv(n, 256), 256, 0, s>>>(p, v, n);
}
static void launch_pad(const float* src, float* dst, long M, int K, int Ks, hipStream_t s) {
  pad_copy_kernel<<<cdiv(M * (long)Ks, 256), 256, 0, s>>>(src, dst, M, K, Ks);
}
static void launch_pack_a(const float* A, __bf16* Ap, long M, int K, int Ks, hipStream_t s) {
  long total = (M >> 4) * (long)(Ks >> 5) * 512;
  pack_a_kernel<<<cdiv(total, 256), 256, 0, s>>>(A, Ap, K, Ks, total);
}
static void launch_pack_w(const float* W, __bf16* Wp, int K, int N, int Ks, int Ns,
                          hipStream_t s) {
  long total = (long)(Ns >> 4) * (Ks >> 5) * 512;
  pack_w_kernel<<<cdiv(total, 256), 256, 0, s>>>(W, Wp, K, N, Ks, Ns, total);
}
// Ns is always a multiple of 32 here -> tilesN even; NB=4 when divisible.
static void launch_gemm(const __bf16* Ap, const __bf16* Wp,
                        const __bf16* Ap2, const __bf16* Wp2,
                        const float* biasp, float* out, int M, int Ns, int Ks, int Ks2,
                        int relu, hipStream_t s) {
  int tilesN = Ns >> 4;
  if ((tilesN & 3) == 0) {
    long tiles = (long)(M >> 4) * (tilesN >> 2);
    gemm_wmma_kernel<4><<<cdiv(tiles, 8), 256, 0, s>>>(Ap, Wp, Ap2, Wp2, biasp, out,
                                                       M, Ns, Ks, Ks2, relu);
  } else {
    long tiles = (long)(M >> 4) * (tilesN >> 1);
    gemm_wmma_kernel<2><<<cdiv(tiles, 8), 256, 0, s>>>(Ap, Wp, Ap2, Wp2, biasp, out,
                                                       M, Ns, Ks, Ks2, relu);
  }
}

struct Scratch {
  float *deg, *agg, *h1, *h2, *h3, *gat;
  float *al, *ar, *mmax, *denom, *alpha;
  float *pooled, *gcnt, *fc1buf;
  __bf16 *apack1, *apack2;          // packed A-operand staging
  __bf16 *wp_c1a, *wp_c1b, *wp_c2a, *wp_c2b, *wp_c3, *wp_g1, *wp_g2;
  float *bp1, *bp2, *bpg1, *bpg2;
};

static void run_branch(const float* x_raw, int Nn, int Fin, int FinP,
                       const int* src, const int* dst, int E,
                       const int* batch,
                       const float* c1Wl, const float* c1Wr, const float* c1b, int F1, int F1P,
                       const float* c2Wl, const float* c2Wr, const float* c2b, int F2, int F2P,
                       const float* c3W, const float* attl, const float* attr,
                       const float* c3b, int F3, int F3P,
                       const float* g1W, const float* g1b,
                       const float* g2W, const float* g2b,
                       float* bout, const Scratch& ws, hipStream_t s) {
  const int total = E + Nn;

  // weight packing + bias padding (tiny)
  launch_pack_w(c1Wl, ws.wp_c1a, Fin, F1, FinP, F1P, s);
  launch_pack_w(c1Wr, ws.wp_c1b, Fin, F1, FinP, F1P, s);
  launch_pack_w(c2Wl, ws.wp_c2a, F1, F2, F1P, F2P, s);
  launch_pack_w(c2Wr, ws.wp_c2b, F1, F2, F1P, F2P, s);
  launch_pack_w(c3W,  ws.wp_c3,  F2, F3, F2P, F3P, s);
  launch_pack_w(g1W,  ws.wp_g1,  F3, 1024, F3P, 1024, s);
  launch_pack_w(g2W,  ws.wp_g2,  1024, 128, 1024, 128, s);
  launch_pad(c1b, ws.bp1, 1, F1, F1P, s);
  launch_pad(c2b, ws.bp2, 1, F2, F2P, s);
  launch_pad(g1b, ws.bpg1, 1, 1024, 1024, s);
  launch_pad(g2b, ws.bpg2, 1, 128, 128, s);

  // in-degree (same edge list for all layers)
  launch_fill(ws.deg, 0.0f, Nn, s);
  degree_kernel<<<cdiv(E, 256), 256, 0, s>>>(dst, E, ws.deg);

  // SAGE layer 1: h1 = relu(mean(x)@Wl + x@Wr + b)
  launch_fill(ws.agg, 0.0f, (long)Nn * FinP, s);
  edge_scatter_add_kernel<<<E, 128, 0, s>>>(x_raw, src, dst, E, Fin, Fin, FinP, ws.agg);
  divide_by_deg_kernel<<<cdiv((long)Nn * Fin, 256), 256, 0, s>>>(ws.agg, ws.deg, Nn, Fin, FinP);
  launch_pack_a(ws.agg, ws.apack1, Nn, FinP, FinP, s);
  launch_pack_a(x_raw,  ws.apack2, Nn, Fin,  FinP, s);   // pads from raw input
  launch_gemm(ws.apack1, ws.wp_c1a, ws.apack2, ws.wp_c1b, ws.bp1, ws.h1,
              Nn, F1P, FinP, FinP, 1, s);

  // SAGE layer 2: h2 = relu(mean(h1)@Wl + h1@Wr + b)
  launch_fill(ws.agg, 0.0f, (long)Nn * F1P, s);
  edge_scatter_add_kernel<<<E, 128, 0, s>>>(ws.h1, src, dst, E, F1, F1P, F1P, ws.agg);
  divide_by_deg_kernel<<<cdiv((long)Nn * F1, 256), 256, 0, s>>>(ws.agg, ws.deg, Nn, F1, F1P);
  launch_pack_a(ws.agg, ws.apack1, Nn, F1P, F1P, s);
  launch_pack_a(ws.h1,  ws.apack2, Nn, F1P, F1P, s);
  launch_gemm(ws.apack1, ws.wp_c2a, ws.apack2, ws.wp_c2b, ws.bp2, ws.h2,
              Nn, F2P, F1P, F1P, 1, s);

  // SuperGAT projection: h3 = h2 @ W  (zero bias: reuse ws.mmax, F3P <= Nn)
  launch_fill(ws.mmax, 0.0f, F3P, s);
  launch_pack_a(ws.h2, ws.apack1, Nn, F2P, F2P, s);
  launch_gemm(ws.apack1, ws.wp_c3, nullptr, nullptr, ws.mmax, ws.h3,
              Nn, F3P, F2P, 0, 0, s);

  node_att_kernel<<<cdiv(Nn, 4), 128, 0, s>>>(ws.h3, Nn, F3, F3P, attl, attr, ws.al, ws.ar);
  launch_fill(ws.mmax, -3.0e38f, Nn, s);
  launch_fill(ws.denom, 0.0f, Nn, s);
  edge_alpha_kernel<<<cdiv(total, 4), 128, 0, s>>>(ws.h3, src, dst, E, Nn, F3, F3P,
                                                   ws.al, ws.ar, ws.alpha, ws.mmax);
  edge_exp_kernel<<<cdiv(total, 256), 256, 0, s>>>(src, dst, E, Nn, ws.alpha,
                                                   ws.mmax, ws.denom);
  launch_fill(ws.gat, 0.0f, (long)Nn * F3P, s);
  edge_aggregate_kernel<<<total, 128, 0, s>>>(ws.h3, src, dst, E, Nn, F3, F3P,
                                              ws.alpha, ws.denom, ws.gat);
  bias_relu_kernel<<<cdiv((long)Nn * F3, 256), 256, 0, s>>>(ws.gat, c3b, Nn, F3, F3P);

  // global mean pool (padded stride; pad cols stay zero)
  launch_fill(ws.pooled, 0.0f, (long)NUM_GRAPHS * F3P, s);
  launch_fill(ws.gcnt, 0.0f, NUM_GRAPHS, s);
  pool_add_kernel<<<Nn, 128, 0, s>>>(ws.gat, batch, Nn, F3, F3P, ws.pooled, ws.gcnt);
  pool_div_kernel<<<cdiv((long)NUM_GRAPHS * F3, 256), 256, 0, s>>>(ws.pooled, ws.gcnt, F3, F3P);

  // branch FC head
  launch_pack_a(ws.pooled, ws.apack1, NUM_GRAPHS, F3P, F3P, s);
  launch_gemm(ws.apack1, ws.wp_g1, nullptr, nullptr, ws.bpg1, ws.fc1buf,
              NUM_GRAPHS, 1024, F3P, 0, 1, s);
  launch_pack_a(ws.fc1buf, ws.apack1, NUM_GRAPHS, 1024, 1024, s);
  launch_gemm(ws.apack1, ws.wp_g2, nullptr, nullptr, ws.bpg2, bout,
              NUM_GRAPHS, 128, 1024, 0, 0, s);
}

extern "C" void kernel_launch(void* const* d_in, const int* in_sizes, int n_in,
                              void* d_out, int out_size, void* d_ws, size_t ws_size,
                              hipStream_t stream) {
  (void)in_sizes; (void)n_in; (void)out_size; (void)ws_size;
  // --- inputs (setup_inputs dict order) ---
  const float* lig_x   = (const float*)d_in[0];   // 50000 x 78
  const int*   lig_ei  = (const int*)  d_in[1];   // 2 x 400000
  const int*   lig_bt  = (const int*)  d_in[2];   // 50000
  const float* pro_x   = (const float*)d_in[3];   // 100000 x 54
  const int*   pro_ei  = (const int*)  d_in[4];   // 2 x 600000
  const int*   pro_bt  = (const int*)  d_in[5];   // 100000
  // --- params: JAX pytree leaves of the dict (alphabetical key order) ---
  int i = 6;
  const float* fc1_W   = (const float*)d_in[i++]; // 256x1024
  const float* fc1_b   = (const float*)d_in[i++];
  const float* fc2_W   = (const float*)d_in[i++]; // 1024x512
  const float* fc2_b   = (const float*)d_in[i++];
  const float* lig_c1_Wl = (const float*)d_in[i++];
  const float* lig_c1_Wr = (const float*)d_in[i++];
  const float* lig_c1_b  = (const float*)d_in[i++];
  const float* lig_c2_Wl = (const float*)d_in[i++];
  const float* lig_c2_Wr = (const float*)d_in[i++];
  const float* lig_c2_b  = (const float*)d_in[i++];
  const float* lig_c3_W  = (const float*)d_in[i++];
  const float* lig_c3_al = (const float*)d_in[i++];
  const float* lig_c3_ar = (const float*)d_in[i++];
  const float* lig_c3_b  = (const float*)d_in[i++];
  const float* lig_g1_W  = (const float*)d_in[i++];
  const float* lig_g1_b  = (const float*)d_in[i++];
  const float* lig_g2_W  = (const float*)d_in[i++];
  const float* lig_g2_b  = (const float*)d_in[i++];
  const float* out_W     = (const float*)d_in[i++]; // 512x1
  const float* out_b     = (const float*)d_in[i++]; // 1
  const float* pro_c1_Wl = (const float*)d_in[i++];
  const float* pro_c1_Wr = (const float*)d_in[i++];
  const float* pro_c1_b  = (const float*)d_in[i++];
  const float* pro_c2_Wl = (const float*)d_in[i++];
  const float* pro_c2_Wr = (const float*)d_in[i++];
  const float* pro_c2_b  = (const float*)d_in[i++];
  const float* pro_c3_W  = (const float*)d_in[i++];
  const float* pro_c3_al = (const float*)d_in[i++];
  const float* pro_c3_ar = (const float*)d_in[i++];
  const float* pro_c3_b  = (const float*)d_in[i++];
  const float* pro_g1_W  = (const float*)d_in[i++];
  const float* pro_g1_b  = (const float*)d_in[i++];
  const float* pro_g2_W  = (const float*)d_in[i++];
  const float* pro_g2_b  = (const float*)d_in[i++];

  const int LN = 50000,  LE = 400000;
  const int PN = 100000, PE = 600000;
  const int* lig_src = lig_ei;            const int* lig_dst = lig_ei + LE;
  const int* pro_src = pro_ei;            const int* pro_dst = pro_ei + PE;

  // --- workspace bump allocation (branch-shared pool, max over branches) ---
  size_t off = 0;
  auto alloc_bytes = [&](size_t b) -> void* {
    void* p = (void*)((char*)d_ws + off);
    off += (b + 255) & ~(size_t)255;
    return p;
  };
  auto alloc  = [&](size_t n) -> float*  { return (float*)alloc_bytes(n * 4); };
  auto allocb = [&](size_t n) -> __bf16* { return (__bf16*)alloc_bytes(n * 2); };

  Scratch ws;
  ws.deg    = alloc(PN);
  ws.agg    = alloc(6400000);             // max(50000*96, 100000*64)
  ws.h1     = alloc(6400000);
  ws.h2     = alloc(12800000);            // max(50000*160, 100000*128)
  ws.h3     = alloc(22400000);            // max(50000*320, 100000*224)
  ws.gat    = alloc(22400000);
  ws.al     = alloc(PN);
  ws.ar     = alloc(PN);
  ws.mmax   = alloc(PN);
  ws.denom  = alloc(PN);
  ws.alpha  = alloc(PE + PN);             // 700000
  ws.pooled = alloc((size_t)NUM_GRAPHS * 320);
  ws.gcnt   = alloc(NUM_GRAPHS);
  ws.fc1buf = alloc((size_t)NUM_GRAPHS * 1024);
  ws.apack1 = allocb(12800000);           // largest A panel (h2)
  ws.apack2 = allocb(6400000);
  ws.wp_c1a = allocb(96 * 96);
  ws.wp_c1b = allocb(96 * 96);
  ws.wp_c2a = allocb(96 * 160);
  ws.wp_c2b = allocb(96 * 160);
  ws.wp_c3  = allocb(160 * 320);
  ws.wp_g1  = allocb(320 * 1024);
  ws.wp_g2  = allocb(1024 * 128);
  ws.bp1    = alloc(96);
  ws.bp2    = alloc(160);
  ws.bpg1   = alloc(1024);
  ws.bpg2   = alloc(128);
  __bf16* wp_fc1 = allocb(256 * 1024);
  __bf16* wp_fc2 = allocb(1024 * 512);
  __bf16* wp_out = allocb(512 * 32);
  float* bpf1 = alloc(1024);
  float* bpf2 = alloc(512);
  float* bpo  = alloc(32);
  float* boutL = alloc((size_t)NUM_GRAPHS * 128);
  float* boutP = alloc((size_t)NUM_GRAPHS * 128);
  float* cat   = alloc((size_t)NUM_GRAPHS * 256);
  float* j1    = alloc((size_t)NUM_GRAPHS * 1024);
  float* j2    = alloc((size_t)NUM_GRAPHS * 512);
  float* j3    = alloc((size_t)NUM_GRAPHS * 32);

  // --- ligand branch (Fin 78->96, F1 78->96, F2 156->160, F3 312->320) ---
  run_branch(lig_x, LN, 78, 96, lig_src, lig_dst, LE, lig_bt,
             lig_c1_Wl, lig_c1_Wr, lig_c1_b, 78, 96,
             lig_c2_Wl, lig_c2_Wr, lig_c2_b, 156, 160,
             lig_c3_W, lig_c3_al, lig_c3_ar, lig_c3_b, 312, 320,
             lig_g1_W, lig_g1_b, lig_g2_W, lig_g2_b,
             boutL, ws, stream);
  // --- protein branch (54->64, 54->64, 108->128, 216->224) ---
  run_branch(pro_x, PN, 54, 64, pro_src, pro_dst, PE, pro_bt,
             pro_c1_Wl, pro_c1_Wr, pro_c1_b, 54, 64,
             pro_c2_Wl, pro_c2_Wr, pro_c2_b, 108, 128,
             pro_c3_W, pro_c3_al, pro_c3_ar, pro_c3_b, 216, 224,
             pro_g1_W, pro_g1_b, pro_g2_W, pro_g2_b,
             boutP, ws, stream);

  // --- joint head ---
  concat_kernel<<<cdiv((long)NUM_GRAPHS * 128, 256), 256, 0, stream>>>(
      boutL, boutP, cat, NUM_GRAPHS, 128);
  launch_pack_w(fc1_W, wp_fc1, 256, 1024, 256, 1024, stream);
  launch_pack_w(fc2_W, wp_fc2, 1024, 512, 1024, 512, stream);
  launch_pack_w(out_W, wp_out, 512, 1, 512, 32, stream);
  launch_pad(fc1_b, bpf1, 1, 1024, 1024, stream);
  launch_pad(fc2_b, bpf2, 1, 512, 512, stream);
  launch_pad(out_b, bpo, 1, 1, 32, stream);
  launch_pack_a(cat, ws.apack1, NUM_GRAPHS, 256, 256, stream);
  launch_gemm(ws.apack1, wp_fc1, nullptr, nullptr, bpf1, j1,
              NUM_GRAPHS, 1024, 256, 0, 1, stream);
  launch_pack_a(j1, ws.apack1, NUM_GRAPHS, 1024, 1024, stream);
  launch_gemm(ws.apack1, wp_fc2, nullptr, nullptr, bpf2, j2,
              NUM_GRAPHS, 512, 1024, 0, 1, stream);
  launch_pack_a(j2, ws.apack1, NUM_GRAPHS, 512, 512, stream);
  launch_gemm(ws.apack1, wp_out, nullptr, nullptr, bpo, j3,
              NUM_GRAPHS, 32, 512, 0, 0, stream);
  copy_col0_kernel<<<cdiv(NUM_GRAPHS, 256), 256, 0, stream>>>(j3, (float*)d_out,
                                                              NUM_GRAPHS, 32);
}
